// LHC_Module_69827578298584
// MI455X (gfx1250) — matrix operand order repaired and checked by reading, not compiled, not run on hardware
//
#include <hip/hip_runtime.h>
#include <hip/hip_bf16.h>

// Problem constants
#define BB   64
#define RR   28
#define PR   30              // padded spatial extent (28 + 1 halo each side)
#define CC   256
#define HH   8
#define EE   128
#define DRES 98              // R*R/H
#define SS   (RR*RR)         // 784
#define PS   (PR*PR)         // 900
#define DPAD 128             // DRES padded to K-multiple-of-32
#define NPAD 112             // DRES padded to N-multiple-of-16

typedef __attribute__((ext_vector_type(16))) __bf16 bf16x16;
typedef __attribute__((ext_vector_type(8)))  float  f32x8;

union Frag { bf16x16 v; uint4 q[2]; };

__device__ __forceinline__ f32x8 wmma_bf16(bf16x16 a, bf16x16 b, f32x8 c) {
  return __builtin_amdgcn_wmma_f32_16x16x32_bf16(false, a, false, b, (short)0, c, false, false);
}

// A-matrix 16x32 bf16 fragment (row-major source, Kpad multiple of 32):
// lane<16 : elems 0..7 = K kbase+0..7,  elems 8..15 = K kbase+16..23
// lane>=16: elems 0..7 = K kbase+8..15, elems 8..15 = K kbase+24..31
__device__ __forceinline__ bf16x16 load_a_frag(const __bf16* row, int kbase, bool hi) {
  Frag f;
  const __bf16* p = row + kbase + (hi ? 8 : 0);
  f.q[0] = *reinterpret_cast<const uint4*>(p);
  f.q[1] = *reinterpret_cast<const uint4*>(p + 16);
  return f.v;
}

// B-matrix 32x16 bf16 fragment from B^T (row = column n of B, contiguous K):
// lane<16 : elems 0..7 = K kbase+0..7, 8..15 = K kbase+16..23 ; lane>=16 : elems 0..7 = K kbase+8..15, 8..15 = K kbase+24..31
__device__ __forceinline__ bf16x16 load_b_frag(const __bf16* row, int kbase, bool hi) {
  Frag f;
  const __bf16* p = row + kbase + (hi ? 8 : 0);
  f.q[0] = *reinterpret_cast<const uint4*>(p);
  f.q[1] = *reinterpret_cast<const uint4*>(p + 16);
  return f.v;
}

// ---------------------------------------------------------------------------
// K-1: zero-fill the padded bf16 image buffer (uint4 stores)
// ---------------------------------------------------------------------------
__global__ void k_zero(uint4* __restrict__ p, int n16) {
  int tid = blockIdx.x * blockDim.x + threadIdx.x;
  if (tid < n16) p[tid] = make_uint4(0u, 0u, 0u, 0u);
}

// ---------------------------------------------------------------------------
// K0: weights -> bf16, transposed (N,K) layouts, zero-padded K for wqk
//   wvt [tap][cout][cin]           (9,256,256)
//   wqkt[h][e][dpad]               (8,128,128), d>=98 -> 0
// ---------------------------------------------------------------------------
__global__ void k_prep(const float* __restrict__ wv_w, const float* __restrict__ wqk_w,
                       __bf16* __restrict__ wvt, __bf16* __restrict__ wqkt) {
  int tid = blockIdx.x * blockDim.x + threadIdx.x;
  const int n_wvt = 9 * CC * CC;
  if (tid < n_wvt) {
    int tap = tid / (CC * CC);
    int rem = tid % (CC * CC);
    int cout = rem / CC, cin = rem % CC;
    float w = wv_w[(tap * CC + cin) * CC + cout];        // [3,3,Cin,Cout]
    wvt[(tap * CC + cout) * CC + cin] = (__bf16)w;
  } else {
    int t = tid - n_wvt;
    if (t < HH * EE * DPAD) {
      int h = t / (EE * DPAD);
      int rem = t % (EE * DPAD);
      int e = rem / DPAD, d = rem % DPAD;
      float w = (d < DRES) ? wqk_w[(h * DRES + d) * EE + e] : 0.f;
      wqkt[(h * EE + e) * DPAD + d] = (__bf16)w;
    }
  }
}

// ---------------------------------------------------------------------------
// K1: avg/max 3x3 SAME pools -> head layout bf16 [b][h][c][DPAD] (pad=0),
//     plus bf16 copy of x into the spatially padded buffer xpad[b][30][30][c]
//     (interior only; border pre-zeroed by k_zero).
// ---------------------------------------------------------------------------
__global__ void k_pool(const float* __restrict__ x, __bf16* __restrict__ xpad,
                       __bf16* __restrict__ qp, __bf16* __restrict__ kp) {
  int tid = blockIdx.x * blockDim.x + threadIdx.x;
  const int total = BB * HH * CC * DPAD;
  if (tid >= total) return;
  int d  = tid % DPAD;
  int ch = (tid / DPAD) % CC;
  int h  = (tid / (DPAD * CC)) % HH;
  int b  =  tid / (DPAD * CC * HH);
  long o = (((long)b * HH + h) * CC + ch) * DPAD + d;
  if (d >= DRES) { qp[o] = (__bf16)0.f; kp[o] = (__bf16)0.f; return; }
  int s = h * DRES + d;
  int r = s / RR, cc = s % RR;
  float sum = 0.f, mx = -3.4e38f, center = 0.f;
  int cnt = 0;
  for (int dy = -1; dy <= 1; ++dy)
    for (int dx = -1; dx <= 1; ++dx) {
      int rr = r + dy, c2 = cc + dx;
      if (rr >= 0 && rr < RR && c2 >= 0 && c2 < RR) {
        float v = x[((long)b * SS + rr * RR + c2) * CC + ch];
        sum += v; ++cnt;
        mx = fmaxf(mx, v);
        if (dy == 0 && dx == 0) center = v;
      }
    }
  qp[o] = (__bf16)(sum / (float)cnt);
  kp[o] = (__bf16)mx;
  xpad[(((long)b * PR + (r + 1)) * PR + (cc + 1)) * CC + ch] = (__bf16)center;
}

// ---------------------------------------------------------------------------
// K2: 3x3 SAME conv as implicit GEMM on the zero-padded image. One wave per
//     16(pixels) x 64(cout) tile: 4 accumulator chains, full A-row (8 frags)
//     hoisted per tap -> 32 WMMAs per A fetch, no predication in hot loop.
// ---------------------------------------------------------------------------
__global__ __launch_bounds__(32) void k_conv(const __bf16* __restrict__ xpad,
                                             const __bf16* __restrict__ wvt,
                                             const float* __restrict__ wv_b,
                                             float* __restrict__ vconv) {
  int idx = blockIdx.x;                 // BB * 49 * 4
  int ng = idx & 3;                     // couts [ng*64, ng*64+64)
  int mt = (idx >> 2) % 49;
  int b  = idx / (4 * 49);
  int lane = threadIdx.x;
  bool hi = lane >= 16;
  int lm = lane & 15;
  int m = mt * 16 + lm;                 // this lane's A-row pixel
  int r = m / RR, cc = m % RR;
  const __bf16* abase = xpad + (((long)b * PR + (r + 1)) * PR + (cc + 1)) * CC;
  int ncol[4];
  const __bf16* bbase[4];
  #pragma unroll
  for (int j = 0; j < 4; ++j) {
    ncol[j] = ng * 64 + j * 16 + lm;
    bbase[j] = wvt + (long)ncol[j] * CC;
  }
  f32x8 acc[4] = {};
  for (int tap = 0; tap < 9; ++tap) {
    int dy = tap / 3 - 1, dx = tap % 3 - 1;
    const __bf16* arow = abase + (long)(dy * PR + dx) * CC;
    bf16x16 a[8];
    #pragma unroll
    for (int ks = 0; ks < 8; ++ks) a[ks] = load_a_frag(arow, ks * 32, hi);
    long boff = (long)tap * CC * CC;
    #pragma unroll
    for (int ks = 0; ks < 8; ++ks) {
      #pragma unroll
      for (int j = 0; j < 4; ++j)
        acc[j] = wmma_bf16(a[ks], load_b_frag(bbase[j] + boff, ks * 32, hi), acc[j]);
    }
  }
  #pragma unroll
  for (int j = 0; j < 4; ++j) {
    float bias = wv_b[ncol[j]];
    #pragma unroll
    for (int i = 0; i < 8; ++i) {
      int mr = mt * 16 + (hi ? i + 8 : i);
      vconv[((long)b * SS + mr) * CC + ncol[j]] = acc[j][i] + bias;
    }
  }
}

// ---------------------------------------------------------------------------
// K3: avg-pool conv output, store TRANSPOSED head layout vT[b][h][d][f] bf16
//     (this is B^T for the final att GEMM), d padded to 112 with zeros.
// ---------------------------------------------------------------------------
__global__ void k_vpool(const float* __restrict__ vconv, __bf16* __restrict__ vT) {
  int tid = blockIdx.x * blockDim.x + threadIdx.x;
  const int total = BB * HH * NPAD * CC;
  if (tid >= total) return;
  int f = tid % CC;
  int d = (tid / CC) % NPAD;
  int h = (tid / (CC * NPAD)) % HH;
  int b =  tid / (CC * NPAD * HH);
  if (d >= DRES) { vT[tid] = (__bf16)0.f; return; }
  int s = h * DRES + d;
  int r = s / RR, cc = s % RR;
  float sum = 0.f; int cnt = 0;
  for (int dy = -1; dy <= 1; ++dy)
    for (int dx = -1; dx <= 1; ++dx) {
      int rr = r + dy, c2 = cc + dx;
      if (rr >= 0 && rr < RR && c2 >= 0 && c2 < RR) {
        sum += vconv[((long)b * SS + rr * RR + c2) * CC + f];
        ++cnt;
      }
    }
  vT[tid] = (__bf16)(sum / (float)cnt);
}

// ---------------------------------------------------------------------------
// K4: per-head dense for q and k: [256 x DPAD] @ wqkt^T -> [256 x 128] bf16
//     A cached across 8 N-tiles; K chain split into 2 accumulators.
// ---------------------------------------------------------------------------
__global__ __launch_bounds__(32) void k_qk(const __bf16* __restrict__ qp,
                                           const __bf16* __restrict__ kp,
                                           const __bf16* __restrict__ wqkt,
                                           const float* __restrict__ wqk_b,
                                           __bf16* __restrict__ qe, __bf16* __restrict__ ke) {
  int idx = blockIdx.x;                 // 2 * (BB*HH) * 16
  int mt  = idx % 16;
  int bh  = (idx / 16) % (BB * HH);
  int isK = idx / (16 * BB * HH);
  const __bf16* src = isK ? kp : qp;
  __bf16* dst = isK ? ke : qe;
  int h = bh & 7;
  int lane = threadIdx.x;
  bool hi = lane >= 16;
  int lm = lane & 15;
  int c = mt * 16 + lm;
  const __bf16* arow = src + ((long)bh * CC + c) * DPAD;
  bf16x16 a[4];
  #pragma unroll
  for (int ks = 0; ks < 4; ++ks) a[ks] = load_a_frag(arow, ks * 32, hi);
  for (int nt = 0; nt < 8; ++nt) {
    int n = nt * 16 + lm;
    const __bf16* brow = wqkt + ((long)h * EE + n) * DPAD;
    f32x8 acc0 = {}, acc1 = {};
    acc0 = wmma_bf16(a[0], load_b_frag(brow,  0, hi), acc0);
    acc1 = wmma_bf16(a[1], load_b_frag(brow, 32, hi), acc1);
    acc0 = wmma_bf16(a[2], load_b_frag(brow, 64, hi), acc0);
    acc1 = wmma_bf16(a[3], load_b_frag(brow, 96, hi), acc1);
    f32x8 acc = acc0 + acc1;
    float bias = wqk_b[h * EE + n];
    #pragma unroll
    for (int i = 0; i < 8; ++i) {
      int cr = mt * 16 + (hi ? i + 8 : i);
      dst[((long)bh * CC + cr) * EE + n] = (__bf16)(acc[i] + bias);
    }
  }
}

// ---------------------------------------------------------------------------
// K5a: scores = qe @ ke^T  (ke row-major == B^T), fp32 out [bh][c][f]
// ---------------------------------------------------------------------------
__global__ __launch_bounds__(32) void k_scores(const __bf16* __restrict__ qe,
                                               const __bf16* __restrict__ ke,
                                               float* __restrict__ scores) {
  int idx = blockIdx.x;                 // (BB*HH) * 16
  int mt = idx % 16;
  int bh = idx / 16;
  int lane = threadIdx.x;
  bool hi = lane >= 16;
  int lm = lane & 15;
  int c = mt * 16 + lm;
  const __bf16* arow = qe + ((long)bh * CC + c) * EE;
  bf16x16 a[4];
  #pragma unroll
  for (int ks = 0; ks < 4; ++ks) a[ks] = load_a_frag(arow, ks * 32, hi);
  for (int nt = 0; nt < 16; ++nt) {
    int f = nt * 16 + lm;
    const __bf16* brow = ke + ((long)bh * CC + f) * EE;
    f32x8 acc0 = {}, acc1 = {};
    acc0 = wmma_bf16(a[0], load_b_frag(brow,  0, hi), acc0);
    acc1 = wmma_bf16(a[1], load_b_frag(brow, 32, hi), acc1);
    acc0 = wmma_bf16(a[2], load_b_frag(brow, 64, hi), acc0);
    acc1 = wmma_bf16(a[3], load_b_frag(brow, 96, hi), acc1);
    f32x8 acc = acc0 + acc1;
    #pragma unroll
    for (int i = 0; i < 8; ++i) {
      int cr = mt * 16 + (hi ? i + 8 : i);
      scores[((long)bh * CC + cr) * CC + f] = acc[i];
    }
  }
}

// K5b1: row means of scores -> sm[bh*256 + c]
__global__ void k_rowmean(const float* __restrict__ scores, float* __restrict__ sm) {
  int tid = blockIdx.x * blockDim.x + threadIdx.x;
  if (tid >= BB * HH * CC) return;
  const float4* row = reinterpret_cast<const float4*>(scores + (long)tid * CC);
  float s = 0.f;
  for (int j = 0; j < CC / 4; ++j) {
    float4 v = row[j];
    s += v.x + v.y + v.z + v.w;
  }
  sm[tid] = s * (1.f / (float)CC);
}

// K5b2: dscale = 1 / E^(0.2 + sigmoid(sm @ wp_w + wp_b)) = exp2(-7*(0.2+p))
__global__ void k_temp(const float* __restrict__ sm, const float* __restrict__ wp_w,
                       const float* __restrict__ wp_b, float* __restrict__ dscale) {
  int tid = blockIdx.x * blockDim.x + threadIdx.x;
  if (tid >= BB * HH * CC) return;
  int bh = tid / CC, cp = tid % CC;
  const float* v = sm + (long)bh * CC;
  float acc = wp_b[cp];
  for (int c = 0; c < CC; ++c) acc += v[c] * wp_w[c * CC + cp];
  float p = 1.f / (1.f + __expf(-acc));
  dscale[tid] = exp2f(-7.f * (0.2f + p));
}

// K5c: row softmax of (scores * dscale) -> bf16 A-matrix wA[bh][c][f]
__global__ __launch_bounds__(32) void k_softmax(const float* __restrict__ scores,
                                                const float* __restrict__ dscale,
                                                __bf16* __restrict__ wA) {
  int row = blockIdx.x;                 // BB*HH*CC rows
  int lane = threadIdx.x;
  float inv = dscale[row];
  const float* p = scores + (long)row * CC + lane * 8;
  float4 f0 = *reinterpret_cast<const float4*>(p);
  float4 f1 = *reinterpret_cast<const float4*>(p + 4);
  float v[8] = {f0.x * inv, f0.y * inv, f0.z * inv, f0.w * inv,
                f1.x * inv, f1.y * inv, f1.z * inv, f1.w * inv};
  float mx = v[0];
  for (int j = 1; j < 8; ++j) mx = fmaxf(mx, v[j]);
  for (int m = 1; m < 32; m <<= 1) mx = fmaxf(mx, __shfl_xor(mx, m, 32));
  float s = 0.f;
  for (int j = 0; j < 8; ++j) { v[j] = __expf(v[j] - mx); s += v[j]; }
  for (int m = 1; m < 32; m <<= 1) s += __shfl_xor(s, m, 32);
  float r = 1.f / s;
  union { __bf16 o[8]; uint4 q; } u;
  for (int j = 0; j < 8; ++j) u.o[j] = (__bf16)(v[j] * r);
  *reinterpret_cast<uint4*>(wA + (long)row * CC + lane * 8) = u.q;
}

// ---------------------------------------------------------------------------
// K5d: att = wA @ v  (vT is B^T), fused residual add back to NHWC.
//     Full K row of A (8 frags) cached once; 7 N-tiles stream against it.
// ---------------------------------------------------------------------------
__global__ __launch_bounds__(32) void k_att(const __bf16* __restrict__ wA,
                                            const __bf16* __restrict__ vT,
                                            const float* __restrict__ x,
                                            float* __restrict__ out) {
  int idx = blockIdx.x;                 // (BB*HH) * 16
  int mt = idx % 16;
  int bh = idx / 16;
  int b = bh >> 3, h = bh & 7;
  int lane = threadIdx.x;
  bool hi = lane >= 16;
  int lm = lane & 15;
  int c = mt * 16 + lm;
  const __bf16* arow = wA + ((long)bh * CC + c) * CC;
  bf16x16 a[8];
  #pragma unroll
  for (int ks = 0; ks < 8; ++ks) a[ks] = load_a_frag(arow, ks * 32, hi);
  for (int nt = 0; nt < 7; ++nt) {
    int d = nt * 16 + lm;
    const __bf16* brow = vT + ((long)bh * NPAD + d) * CC;
    f32x8 acc0 = {}, acc1 = {};
    #pragma unroll
    for (int ks = 0; ks < 4; ++ks) {
      acc0 = wmma_bf16(a[2 * ks],     load_b_frag(brow, 2 * ks * 32,       hi), acc0);
      acc1 = wmma_bf16(a[2 * ks + 1], load_b_frag(brow, (2 * ks + 1) * 32, hi), acc1);
    }
    f32x8 acc = acc0 + acc1;
    if (d < DRES) {
      int s = h * DRES + d;
      #pragma unroll
      for (int i = 0; i < 8; ++i) {
        int cr = mt * 16 + (hi ? i + 8 : i);
        long o = ((long)b * SS + s) * CC + cr;
        out[o] = x[o] + acc[i];
      }
    }
  }
}

// ---------------------------------------------------------------------------
extern "C" void kernel_launch(void* const* d_in, const int* in_sizes, int n_in,
                              void* d_out, int out_size, void* d_ws, size_t ws_size,
                              hipStream_t stream) {
  (void)in_sizes; (void)n_in; (void)out_size; (void)ws_size;
  const float* x     = (const float*)d_in[0];
  const float* wqk_w = (const float*)d_in[1];
  const float* wqk_b = (const float*)d_in[2];
  const float* wp_w  = (const float*)d_in[3];
  const float* wp_b  = (const float*)d_in[4];
  const float* wv_w  = (const float*)d_in[5];
  const float* wv_b  = (const float*)d_in[6];
  float* out = (float*)d_out;

  char* ws = (char*)d_ws;
  auto carve = [&](size_t bytes) {
    char* p = ws;
    ws += (bytes + 255) & ~(size_t)255;
    return p;
  };
  __bf16* xpad   = (__bf16*)carve((size_t)BB * PS * CC * 2);
  __bf16* qp     = (__bf16*)carve((size_t)BB * HH * CC * DPAD * 2);
  __bf16* kp     = (__bf16*)carve((size_t)BB * HH * CC * DPAD * 2);
  __bf16* wvt    = (__bf16*)carve((size_t)9 * CC * CC * 2);
  __bf16* wqkt   = (__bf16*)carve((size_t)HH * EE * DPAD * 2);
  float*  vconv  = (float*) carve((size_t)BB * SS * CC * 4);
  __bf16* vT     = (__bf16*)carve((size_t)BB * HH * NPAD * CC * 2);
  __bf16* qe     = (__bf16*)carve((size_t)BB * HH * CC * EE * 2);
  __bf16* ke     = (__bf16*)carve((size_t)BB * HH * CC * EE * 2);
  float*  scores = (float*) carve((size_t)BB * HH * CC * CC * 4);
  float*  smbuf  = (float*) carve((size_t)BB * HH * CC * 4);
  float*  dscale = (float*) carve((size_t)BB * HH * CC * 4);
  __bf16* wAtt   = (__bf16*)carve((size_t)BB * HH * CC * CC * 2);

  {
    int n16 = (int)((size_t)BB * PS * CC * 2 / 16);
    k_zero<<<(n16 + 255) / 256, 256, 0, stream>>>((uint4*)xpad, n16);
  }
  {
    int n = 9 * CC * CC + HH * EE * DPAD;
    k_prep<<<(n + 255) / 256, 256, 0, stream>>>(wv_w, wqk_w, wvt, wqkt);
  }
  {
    int n = BB * HH * CC * DPAD;
    k_pool<<<(n + 255) / 256, 256, 0, stream>>>(x, xpad, qp, kp);
  }
  k_conv<<<BB * 49 * 4, 32, 0, stream>>>(xpad, wvt, wv_b, vconv);
  {
    int n = BB * HH * NPAD * CC;
    k_vpool<<<(n + 255) / 256, 256, 0, stream>>>(vconv, vT);
  }
  k_qk<<<2 * BB * HH * 16, 32, 0, stream>>>(qp, kp, wqkt, wqk_b, qe, ke);
  k_scores<<<BB * HH * 16, 32, 0, stream>>>(qe, ke, scores);
  {
    int n = BB * HH * CC;
    k_rowmean<<<(n + 255) / 256, 256, 0, stream>>>(scores, smbuf);
    k_temp<<<(n + 255) / 256, 256, 0, stream>>>(smbuf, wp_w, wp_b, dscale);
  }
  k_softmax<<<BB * HH * CC, 32, 0, stream>>>(scores, dscale, wAtt);
  k_att<<<BB * HH * 16, 32, 0, stream>>>(wAtt, vT, x, out);
}
